// sLSTMCell_11132555231846
// MI455X (gfx1250) — matrix operand order repaired and checked by reading, not compiled, not run on hardware
//
#include <hip/hip_runtime.h>
#include <hip/hip_bf16.h>

typedef __bf16  bf16;
typedef bf16    bf16x2  __attribute__((ext_vector_type(2)));
typedef bf16    bf16x4  __attribute__((ext_vector_type(4)));
typedef bf16    bf16x8  __attribute__((ext_vector_type(8)));
typedef bf16    bf16x16 __attribute__((ext_vector_type(16)));
typedef float   f32x4   __attribute__((ext_vector_type(4)));
typedef float   f32x8   __attribute__((ext_vector_type(8)));
typedef int     gvi32x4 __attribute__((__vector_size__(16)));   // GCC-style, matches builtin sig

#define B_ROWS 16384
#define IN_DIM 512
#define H_DIM  1024
#define K_DIM  1536   /* IN + H */
#define NG     4096   /* 4*H    */

#define BM 128        /* batch rows per workgroup             */
#define BH 32         /* h-columns per workgroup (x4 gates)   */
#define BK 32         /* K-step per WMMA                      */
#define NKS (K_DIM / BK)            /* 48 K-steps            */
#define LDST 40       /* padded LDS row stride (elements); 80B = 5x16B aligned */
#define TILE_B_BYTES (128 * LDST * 2)   /* 10240 B per prepacked B tile */

// ---------------------------------------------------------------------------
// CDNA5 async copy global->LDS (ASYNCcnt path), builtin if present else asm
// ---------------------------------------------------------------------------
__device__ __forceinline__ void async_copy_b128(const void* g, void* l) {
#if __has_builtin(__builtin_amdgcn_global_load_async_to_lds_b128)
    __builtin_amdgcn_global_load_async_to_lds_b128(
        (__attribute__((address_space(1))) gvi32x4*)(void*)g,
        (__attribute__((address_space(3))) gvi32x4*)l, 0, 0);
#else
    unsigned lds_off = (unsigned)(unsigned long long)l;  // flat low 32b == LDS byte addr
    asm volatile("global_load_async_to_lds_b128 %0, %1, off"
                 :: "v"(lds_off), "v"(g) : "memory");
#endif
}

__device__ __forceinline__ void wait_async0() {
#if __has_builtin(__builtin_amdgcn_s_wait_asynccnt)
    __builtin_amdgcn_s_wait_asynccnt(0);
#else
    asm volatile("s_wait_asynccnt 0x0" ::: "memory");
#endif
}

// ---------------------------------------------------------------------------
// Pass 1a: A = [input | h_t_p] as bf16, row-major [16384 x 1536]
// ---------------------------------------------------------------------------
__global__ __launch_bounds__(256) void pack_a_kernel(
    const float* __restrict__ x, const float* __restrict__ h_p,
    bf16* __restrict__ wsA)
{
    const int row = blockIdx.x;
    const int tid = threadIdx.x;
#pragma unroll
    for (int i = 0; i < 6; ++i) {
        const int k = tid + i * 256;           // 0..1535
        const float v = (k < IN_DIM) ? x[(size_t)row * IN_DIM + k]
                                     : h_p[(size_t)row * H_DIM + (k - IN_DIM)];
        wsA[(size_t)row * K_DIM + k] = (bf16)v;
    }
}

// ---------------------------------------------------------------------------
// Pass 1b: B tiles: for each (h-block hb, k-block kb) a 128col x 32k bf16
// tile, gate-gathered, transposed [col][k], stride LDST — the exact LDS image.
// ---------------------------------------------------------------------------
__global__ __launch_bounds__(256) void pack_b_kernel(
    const float* __restrict__ w_w, const float* __restrict__ r_w,
    bf16* __restrict__ wsB)
{
    const int hb = blockIdx.x;                 // 0..31
    const int kb = blockIdx.y;                 // 0..47
    const int tid = threadIdx.x;
    bf16* tile = wsB + ((size_t)hb * NKS + kb) * (128 * LDST);
#pragma unroll
    for (int i = 0; i < 16; ++i) {
        const int idx = tid + i * 256;         // 0..4095
        const int c   = idx & 127;             // tile column
        const int k   = idx >> 7;              // 0..31
        const int g   = c >> 5;                // gate
        const int gcol = g * H_DIM + hb * BH + (c & 31);
        const int krow = kb * BK + k;
        const float v = (krow < IN_DIM) ? w_w[(size_t)krow * NG + gcol]
                                        : r_w[(size_t)(krow - IN_DIM) * NG + gcol];
        tile[c * LDST + k] = (bf16)v;
    }
}

// ---------------------------------------------------------------------------
// Pass 2: GEMM (bf16 WMMA, async double-buffered LDS) + fused sLSTM epilogue
// ---------------------------------------------------------------------------
__global__ __launch_bounds__(256) void slstm_wmma_async_kernel(
    const bf16*  __restrict__ wsA,   // [16384 x 1536] bf16
    const bf16*  __restrict__ wsB,   // [32][48] prepacked tiles
    const float* __restrict__ c_p,
    const float* __restrict__ n_p,
    const float* __restrict__ m_p,
    const float* __restrict__ w_b,
    const float* __restrict__ r_b,
    float* __restrict__ out)
{
    __shared__ bf16 lA[2][BM * LDST];    // 2 x 10240 B
    __shared__ bf16 lB[2][128 * LDST];   // 2 x 10240 B

    const int tid  = threadIdx.x;
    const int lane = tid & 31;
    const int wave = tid >> 5;
    const int m0   = blockIdx.x * BM;
    const int hb   = blockIdx.y;
    const int h0   = hb * BH;

    const int ll  = lane & 15;
    const int lhi = lane >> 4;

    f32x8 acc[8];
#pragma unroll
    for (int t = 0; t < 8; ++t) acc[t] = (f32x8)0.f;

    // per-thread async chunk bases (pointer-bump only inside the loop)
    const int ca_row0 =  tid        >> 2, ca_part0 =  tid        & 3;
    const int ca_row1 = (tid + 256) >> 2, ca_part1 = (tid + 256) & 3;
    const char* gA0 = (const char*)wsA + (size_t)(m0 + ca_row0) * (K_DIM * 2) + ca_part0 * 16;
    const char* gA1 = (const char*)wsA + (size_t)(m0 + ca_row1) * (K_DIM * 2) + ca_part1 * 16;
    const int   lA0 = ca_row0 * (LDST * 2) + ca_part0 * 16;
    const int   lA1 = ca_row1 * (LDST * 2) + ca_part1 * 16;
    const char* gB0 = (const char*)wsB + (size_t)hb * NKS * TILE_B_BYTES;

    auto issue_stage = [&](int ks, int buf) {
        const char* ga = gA0 + (size_t)ks * (BK * 2);
        async_copy_b128(ga, (char*)&lA[buf][0] + lA0);
        const char* gb = gA1 + (size_t)ks * (BK * 2);
        async_copy_b128(gb, (char*)&lA[buf][0] + lA1);
        const char* gt = gB0 + (size_t)ks * TILE_B_BYTES;
        async_copy_b128(gt + tid * 16,         (char*)&lB[buf][0] + tid * 16);
        async_copy_b128(gt + (tid + 256) * 16, (char*)&lB[buf][0] + (tid + 256) * 16);
        if (tid < 128)
            async_copy_b128(gt + (tid + 512) * 16, (char*)&lB[buf][0] + (tid + 512) * 16);
    };

    issue_stage(0, 0);

    for (int ks = 0; ks < NKS; ++ks) {
        const int cur = ks & 1;
        wait_async0();                 // this wave's stage-ks chunks are in LDS
        __syncthreads();               // everyone's chunks are in LDS
        if (ks + 1 < NKS) issue_stage(ks + 1, cur ^ 1);  // overlap with WMMA

        // ---- A fragment (16x32 bf16, ISA lane-half K-interleave) ----
        const int arow = wave * 16 + ll;
        const int ak0  = lhi * 8;
        bf16x8 a_lo = *(const bf16x8*)&lA[cur][arow * LDST + ak0];
        bf16x8 a_hi = *(const bf16x8*)&lA[cur][arow * LDST + ak0 + 16];
        bf16x16 afrag;
#pragma unroll
        for (int i = 0; i < 8; ++i) { afrag[i] = a_lo[i]; afrag[8 + i] = a_hi[i]; }

        // ---- 8 B fragments + WMMA ----
        const int bkb = lhi * 8;
#pragma unroll
        for (int t = 0; t < 8; ++t) {
            const int c = t * 16 + ll;
            bf16x8 b_lo = *(const bf16x8*)&lB[cur][c * LDST + bkb];
            bf16x8 b_hi = *(const bf16x8*)&lB[cur][c * LDST + bkb + 16];
            bf16x16 bfrag;
#pragma unroll
            for (int i = 0; i < 8; ++i) { bfrag[i] = b_lo[i]; bfrag[8 + i] = b_hi[i]; }
            acc[t] = __builtin_amdgcn_wmma_f32_16x16x32_bf16(
                false, afrag, false, bfrag, (short)0, acc[t], false, false);
        }
        __syncthreads();               // all reads done before buffer reuse
    }

    // ---- fused sLSTM epilogue (pure per-lane VALU) ----
    const int   rowbase = m0 + wave * 16 + lhi * 8;
    const size_t BHsz   = (size_t)B_ROWS * H_DIM;
    float* out_h  = out;
    float* out_c  = out +     BHsz;
    float* out_h2 = out + 2 * BHsz;
    float* out_n  = out + 3 * BHsz;
    float* out_m  = out + 4 * BHsz;

#pragma unroll
    for (int hi = 0; hi < 2; ++hi) {
        const int col = h0 + hi * 16 + ll;
        const float bz  = w_b[0 * H_DIM + col] + r_b[0 * H_DIM + col];
        const float bi  = w_b[1 * H_DIM + col] + r_b[1 * H_DIM + col];
        const float bfg = w_b[2 * H_DIM + col] + r_b[2 * H_DIM + col];
        const float bo  = w_b[3 * H_DIM + col] + r_b[3 * H_DIM + col];
        const f32x8 az = acc[0 + hi], ai = acc[2 + hi], af = acc[4 + hi], ao = acc[6 + hi];
#pragma unroll
        for (int v = 0; v < 8; ++v) {
            const size_t idx = (size_t)(rowbase + v) * H_DIM + col;
            const float zpre = az[v] + bz;
            const float ipre = ai[v] + bi;
            const float fpre = af[v] + bfg;
            const float opre = ao[v] + bo;

            const float cp = c_p[idx];
            const float np = n_p[idx];
            const float mp = m_p[idx];

            const float e2  = __expf(2.f * zpre);
            const float z_t = 1.f - 2.f / (e2 + 1.f);      // tanh, saturating
            const float i_t = __expf(ipre);                 // unstabilized (faithful)
            const float m_t = fmaxf(fpre + mp, ipre);
            const float si  = __expf(ipre - m_t);
            const float sf  = __expf(fpre + mp - m_t);
            const float c_t = sf * cp + i_t * z_t;
            const float n_t = sf * np + si;
            const float o_t = 1.f / (1.f + __expf(-opre));
            const float h_t = o_t * (c_t / n_t);

            out_h [idx] = h_t;
            out_c [idx] = c_t;
            out_h2[idx] = h_t;
            out_n [idx] = n_t;
            out_m [idx] = m_t;
        }
    }
}

// ---------------------------------------------------------------------------
// Fallback: self-contained single kernel (round-1) if ws too small
// ---------------------------------------------------------------------------
__global__ __launch_bounds__(256) void slstm_wmma_fused_kernel(
    const float* __restrict__ x,   const float* __restrict__ c_p,
    const float* __restrict__ h_p, const float* __restrict__ n_p,
    const float* __restrict__ m_p, const float* __restrict__ w_w,
    const float* __restrict__ w_b, const float* __restrict__ r_w,
    const float* __restrict__ r_b, float* __restrict__ out)
{
    __shared__ bf16 lA[BM * LDST];
    __shared__ bf16 lB[128 * LDST];

    const int tid  = threadIdx.x;
    const int lane = tid & 31;
    const int wave = tid >> 5;
    const int m0   = blockIdx.x * BM;
    const int h0   = blockIdx.y * BH;
    const int ll   = lane & 15;
    const int lhi  = lane >> 4;

    f32x8 acc[8];
#pragma unroll
    for (int t = 0; t < 8; ++t) acc[t] = (f32x8)0.f;

    for (int ks = 0; ks < NKS; ++ks) {
        const int kb = ks * BK;
        {
            const float* src; int stride, koff;
            if (kb < IN_DIM) { src = x;   stride = IN_DIM; koff = kb; }
            else             { src = h_p; stride = H_DIM;  koff = kb - IN_DIM; }
#pragma unroll
            for (int i = 0; i < 4; ++i) {
                const int idx4 = tid + i * 256;
                const int row  = idx4 >> 3;
                const int kc   = (idx4 & 7) * 4;
                f32x4 v = *(const f32x4*)(src + (size_t)(m0 + row) * stride + koff + kc);
                bf16x4 p = { (bf16)v.x, (bf16)v.y, (bf16)v.z, (bf16)v.w };
                *(bf16x4*)&lA[row * LDST + kc] = p;
            }
        }
        {
#pragma unroll
            for (int i = 0; i < 8; ++i) {
                const int idx  = tid + i * 256;
                const int n    = idx & 127;
                const int k    = (idx >> 7) * 2;
                const int g    = n >> 5;
                const int gcol = g * H_DIM + h0 + (n & 31);
                const int krow = kb + k;
                const float* wsrc; int wr;
                if (krow < IN_DIM) { wsrc = w_w; wr = krow; }
                else               { wsrc = r_w; wr = krow - IN_DIM; }
                const float f0 = wsrc[(size_t)wr * NG + gcol];
                const float f1 = wsrc[(size_t)(wr + 1) * NG + gcol];
                bf16x2 p = { (bf16)f0, (bf16)f1 };
                *(bf16x2*)&lB[n * LDST + k] = p;
            }
        }
        __syncthreads();

        const int arow = wave * 16 + ll;
        const int ak0  = lhi * 8;
        bf16x8 a_lo = *(const bf16x8*)&lA[arow * LDST + ak0];
        bf16x8 a_hi = *(const bf16x8*)&lA[arow * LDST + ak0 + 16];
        bf16x16 afrag;
#pragma unroll
        for (int i = 0; i < 8; ++i) { afrag[i] = a_lo[i]; afrag[8 + i] = a_hi[i]; }

        const int bkb = lhi * 8;
#pragma unroll
        for (int t = 0; t < 8; ++t) {
            const int c = t * 16 + ll;
            bf16x8 b_lo = *(const bf16x8*)&lB[c * LDST + bkb];
            bf16x8 b_hi = *(const bf16x8*)&lB[c * LDST + bkb + 16];
            bf16x16 bfrag;
#pragma unroll
            for (int i = 0; i < 8; ++i) { bfrag[i] = b_lo[i]; bfrag[8 + i] = b_hi[i]; }
            acc[t] = __builtin_amdgcn_wmma_f32_16x16x32_bf16(
                false, afrag, false, bfrag, (short)0, acc[t], false, false);
        }
        __syncthreads();
    }

    const int   rowbase = m0 + wave * 16 + lhi * 8;
    const size_t BHsz   = (size_t)B_ROWS * H_DIM;
    float* out_h  = out;
    float* out_c  = out +     BHsz;
    float* out_h2 = out + 2 * BHsz;
    float* out_n  = out + 3 * BHsz;
    float* out_m  = out + 4 * BHsz;

#pragma unroll
    for (int hi = 0; hi < 2; ++hi) {
        const int col = h0 + hi * 16 + ll;
        const float bz  = w_b[0 * H_DIM + col] + r_b[0 * H_DIM + col];
        const float bi  = w_b[1 * H_DIM + col] + r_b[1 * H_DIM + col];
        const float bfg = w_b[2 * H_DIM + col] + r_b[2 * H_DIM + col];
        const float bo  = w_b[3 * H_DIM + col] + r_b[3 * H_DIM + col];
        const f32x8 az = acc[0 + hi], ai = acc[2 + hi], af = acc[4 + hi], ao = acc[6 + hi];
#pragma unroll
        for (int v = 0; v < 8; ++v) {
            const size_t idx = (size_t)(rowbase + v) * H_DIM + col;
            const float zpre = az[v] + bz;
            const float ipre = ai[v] + bi;
            const float fpre = af[v] + bfg;
            const float opre = ao[v] + bo;
            const float cp = c_p[idx], np = n_p[idx], mp = m_p[idx];
            const float e2  = __expf(2.f * zpre);
            const float z_t = 1.f - 2.f / (e2 + 1.f);
            const float i_t = __expf(ipre);
            const float m_t = fmaxf(fpre + mp, ipre);
            const float si  = __expf(ipre - m_t);
            const float sf  = __expf(fpre + mp - m_t);
            const float c_t = sf * cp + i_t * z_t;
            const float n_t = sf * np + si;
            const float o_t = 1.f / (1.f + __expf(-opre));
            const float h_t = o_t * (c_t / n_t);
            out_h [idx] = h_t;
            out_c [idx] = c_t;
            out_h2[idx] = h_t;
            out_n [idx] = n_t;
            out_m [idx] = m_t;
        }
    }
}

extern "C" void kernel_launch(void* const* d_in, const int* in_sizes, int n_in,
                              void* d_out, int out_size, void* d_ws, size_t ws_size,
                              hipStream_t stream) {
    (void)in_sizes; (void)n_in; (void)out_size;
    const float* x    = (const float*)d_in[0];
    const float* c_p  = (const float*)d_in[1];
    const float* h_p  = (const float*)d_in[2];
    const float* n_p  = (const float*)d_in[3];
    const float* m_p  = (const float*)d_in[4];
    const float* w_w  = (const float*)d_in[5];
    const float* w_b  = (const float*)d_in[6];
    const float* r_w  = (const float*)d_in[7];
    const float* r_b  = (const float*)d_in[8];
    float* out = (float*)d_out;

    const size_t needA = (size_t)B_ROWS * K_DIM * sizeof(bf16);          // 48 MiB
    const size_t needB = (size_t)32 * NKS * TILE_B_BYTES;                // 15 MiB
    dim3 grid(B_ROWS / BM, H_DIM / BH);   // 128 x 32
    dim3 block(256);                      // 8 waves (wave32)

    if (ws_size >= needA + needB) {
        bf16* wsA = (bf16*)d_ws;
        bf16* wsB = (bf16*)((char*)d_ws + needA);
        pack_a_kernel<<<dim3(B_ROWS), block, 0, stream>>>(x, h_p, wsA);
        pack_b_kernel<<<dim3(32, NKS), block, 0, stream>>>(w_w, r_w, wsB);
        slstm_wmma_async_kernel<<<grid, block, 0, stream>>>(
            wsA, wsB, c_p, n_p, m_p, w_b, r_b, out);
    } else {
        slstm_wmma_fused_kernel<<<grid, block, 0, stream>>>(
            x, c_p, h_p, n_p, m_p, w_w, w_b, r_w, r_b, out);
    }
}
